// JetMoeMoE_25546465477252
// MI455X (gfx1250) — matrix operand order repaired and checked by reading, not compiled, not run on hardware
//
#include <hip/hip_runtime.h>
#include <hip/hip_bf16.h>
#include <stdint.h>

#define DEVI __device__ __forceinline__

typedef __attribute__((ext_vector_type(16))) __bf16 v16bf;
typedef __attribute__((ext_vector_type(8)))  float  v8f;

// ---------------- problem dims (fixed by reference) ----------------
constexpr int B_   = 2;
constexpr int L_   = 2048;
constexpr int D_   = 2048;
constexpr int H_   = 4096;
constexpr int E_   = 8;
constexpr int TOPK = 2;
constexpr int T_   = B_ * L_;   // 4096 tokens
constexpr int H2_  = 2 * H_;    // 8192

// ---------------- tiling ----------------
constexpr int KT       = 32;    // k-tile (one wmma k-step)
constexpr int LDST     = 40;    // padded LDS row stride in bf16 elements (80B, 16B aligned)
constexpr int TILE_ELE = 128 * LDST;

// ---------------- workspace layout (bytes) ----------------
constexpr size_t OFF_COUNTS = 0;
constexpr size_t OFF_BASE   = 256;
constexpr size_t OFF_TOKIDX = 512;                                   // E_*T_ ints
constexpr size_t OFF_TOKE   = OFF_TOKIDX + (size_t)E_ * T_ * 4;      // 2T ints
constexpr size_t OFF_TOKP   = OFF_TOKE   + (size_t)TOPK * T_ * 4;    // 2T ints
constexpr size_t OFF_TOKG   = OFF_TOKP   + (size_t)TOPK * T_ * 4;    // 2T floats
constexpr size_t OFF_XB     = (OFF_TOKG + (size_t)TOPK * T_ * 4 + 255) & ~(size_t)255;
constexpr size_t OFF_WINB   = OFF_XB    + (size_t)T_ * D_ * 2;       // bf16 x
constexpr size_t OFF_WOUTB  = OFF_WINB  + (size_t)E_ * H2_ * D_ * 2; // bf16 w_in
constexpr size_t OFF_ACT    = OFF_WOUTB + (size_t)E_ * D_ * H_ * 2;  // bf16 w_out
constexpr size_t OFF_OBUF   = OFF_ACT   + (size_t)TOPK * T_ * H_ * 2;// bf16 act
// obuf: TOPK*T_*D_ floats after OFF_OBUF

// ---------------- helpers ----------------
DEVI unsigned short f2bf(float f) {        // round-to-nearest-even
    unsigned u = __float_as_uint(f);
    unsigned r = 0x7FFFu + ((u >> 16) & 1u);
    return (unsigned short)((u + r) >> 16);
}

struct Pack32 { uint4 lo, hi; };
DEVI v16bf as_v16bf(uint4 lo, uint4 hi) {
    Pack32 p{lo, hi};
    return __builtin_bit_cast(v16bf, p);
}

// A fragment (16x32 bf16, MxK): lane<16 -> K 0..15, lane>=16 -> K 16..31
DEVI v16bf lds_frag_a(const unsigned short* s, int mbase, int lane) {
    int m = mbase + (lane & 15);
    int k = (lane < 16) ? 0 : 16;
    const uint4* p0 = (const uint4*)(s + m * LDST + k);
    const uint4* p1 = (const uint4*)(s + m * LDST + k + 8);
    return as_v16bf(*p0, *p1);
}
// B fragment (32x16 bf16, KxN): lane%16 -> N, lane<16 -> K 0..15, lane>=16 -> K 16..31
DEVI v16bf lds_frag_b(const unsigned short* s, int nbase, int lane) {
    int n = nbase + (lane & 15);
    int k = (lane < 16) ? 0 : 16;
    const uint4* p = (const uint4*)(s + n * LDST + k);
    return as_v16bf(p[0], p[1]);
}

DEVI v8f wmma_bf16(v16bf a, v16bf b, v8f c) {
    return __builtin_amdgcn_wmma_f32_16x16x32_bf16(
        false, a, false, b, (short)0, c, false, false);
}

// async global -> LDS copy, 16B per lane, tracked with ASYNCcnt
DEVI void async_cp16(unsigned lds_byte, const void* g) {
    asm volatile("global_load_async_to_lds_b128 %0, %1, off"
                 :: "v"(lds_byte), "v"((unsigned long long)(uintptr_t)g)
                 : "memory");
}
DEVI void wait_async_le4() { asm volatile("s_wait_asynccnt 4" ::: "memory"); }
DEVI void wait_async_0()   { asm volatile("s_wait_asynccnt 0" ::: "memory"); }

DEVI int imin(int a, int b) { return a < b ? a : b; }

// ---------------- kernel: fp32 -> bf16 convert (float4 / ushort4) ----------------
__global__ __launch_bounds__(256)
void cvt_bf16_kernel(const float* __restrict__ in, unsigned short* __restrict__ out,
                     long long n4) {
    for (long long i = (long long)blockIdx.x * blockDim.x + threadIdx.x;
         i < n4; i += (long long)gridDim.x * blockDim.x) {
        float4 v = ((const float4*)in)[i];
        ushort4 o;
        o.x = f2bf(v.x); o.y = f2bf(v.y); o.z = f2bf(v.z); o.w = f2bf(v.w);
        ((ushort4*)out)[i] = o;
    }
}

// ---------------- kernel: router (one wave per token) ----------------
__global__ __launch_bounds__(256)
void router_kernel(const float* __restrict__ x, const float* __restrict__ rw,
                   int* counts, int* tok_idx, int* tok_e, int* tok_p, float* tok_g) {
    int wave = threadIdx.x >> 5, lane = threadIdx.x & 31;
    int t = blockIdx.x * 8 + wave;
    if (t >= T_) return;

    float acc[E_];
#pragma unroll
    for (int e = 0; e < E_; ++e) acc[e] = 0.f;

    const float* xr = x + (size_t)t * D_;
    for (int i = lane; i < D_; i += 32) {
        float xv = xr[i];
#pragma unroll
        for (int e = 0; e < E_; ++e) acc[e] += xv * rw[e * D_ + i];
    }
#pragma unroll
    for (int e = 0; e < E_; ++e)
        for (int off = 16; off > 0; off >>= 1)
            acc[e] += __shfl_xor(acc[e], off, 32);

    if (lane == 0) {
        int e0 = 0; float v0 = acc[0];
#pragma unroll
        for (int e = 1; e < E_; ++e) if (acc[e] > v0) { v0 = acc[e]; e0 = e; }
        int e1 = -1; float v1 = -3.0e38f;
#pragma unroll
        for (int e = 0; e < E_; ++e)
            if (e != e0 && acc[e] > v1) { v1 = acc[e]; e1 = e; }

        float g0 = 1.f / (1.f + __expf(v1 - v0));   // softmax over top-2
        float g1 = 1.f - g0;

        int p0 = atomicAdd(&counts[e0], 1);
        int p1 = atomicAdd(&counts[e1], 1);
        tok_idx[e0 * T_ + p0] = t;
        tok_idx[e1 * T_ + p1] = t;
        tok_e[2 * t]     = e0; tok_p[2 * t]     = p0; tok_g[2 * t]     = g0;
        tok_e[2 * t + 1] = e1; tok_p[2 * t + 1] = p1; tok_g[2 * t + 1] = g1;
    }
}

// ---------------- kernel: 8-way exclusive scan ----------------
__global__ void scan_kernel(const int* counts, int* bases) {
    if (threadIdx.x == 0 && blockIdx.x == 0) {
        int s = 0;
        for (int e = 0; e < E_; ++e) { bases[e] = s; s += counts[e]; }
    }
}

// ---------------- GEMM epilogue helpers ----------------
DEVI void glu_store(unsigned short* __restrict__ act, v8f a, v8f b,
                    int abase, int gmb, int gn, int cnt) {
#pragma unroll
    for (int r = 0; r < 8; ++r) {
        int m = gmb + r;
        if (m < cnt) {
            float av = a[r];
            float s  = av / (1.f + __expf(-av));
            act[(size_t)(abase + m) * H_ + gn] = f2bf(s * b[r]);
        }
    }
}
DEVI void obuf_store(float* __restrict__ obuf, v8f c,
                     int abase, int gmb, int gn, int cnt) {
#pragma unroll
    for (int r = 0; r < 8; ++r) {
        int m = gmb + r;
        if (m < cnt)
            obuf[(size_t)(abase + m) * D_ + gn] = c[r];
    }
}

// ---------------- kernel: GEMM1 (x @ w_in^T) fused with GLU ----------------
// block: 256 threads = 8 waves (4 along M x 2 along N)
// block tile: M=128 rows (gathered tokens), 64 act cols (both GLU halves computed)
__global__ __launch_bounds__(256)
void gemm1_glu_kernel(const unsigned short* __restrict__ xb,
                      const unsigned short* __restrict__ winb,
                      const int* __restrict__ counts, const int* __restrict__ bases,
                      const int* __restrict__ tok_idx,
                      unsigned short* __restrict__ act) {
    int e   = blockIdx.y;
    int cnt = counts[e];
    int m0  = blockIdx.z * 128;
    if (m0 >= cnt) return;
    int n0    = blockIdx.x * 64;
    int abase = bases[e];

    __shared__ unsigned short shA[2 * TILE_ELE];
    __shared__ unsigned short shB[2 * TILE_ELE];

    int tid  = threadIdx.x;
    int lane = tid & 31;
    int wave = tid >> 5;
    int wm   = wave & 3;      // 0..3 (M)
    int wn   = wave >> 2;     // 0..1 (N)

    // per-thread staging assignment: row = tid/2, two 16B chunks
    int row = tid >> 1;
    int cb  = (tid & 1) * 2;                     // chunk base (of 4 per row)
    int ar  = imin(m0 + row, cnt - 1);
    int tok = tok_idx[e * T_ + ar];
    const unsigned short* aRow = xb + (size_t)tok * D_;
    int wr = (row < 64) ? (n0 + row) : (H_ + n0 + (row - 64));   // a-half then b-half
    const unsigned short* bRow = winb + ((size_t)e * H2_ + wr) * D_;

    unsigned ldsA = (unsigned)(uintptr_t)&shA[0] + (unsigned)(row * LDST + cb * 8) * 2u;
    unsigned ldsB = (unsigned)(uintptr_t)&shB[0] + (unsigned)(row * LDST + cb * 8) * 2u;

    auto issue = [&](unsigned tileOff, int k0) {
        unsigned boff = tileOff * 2u;
        const unsigned short* ga = aRow + k0 + cb * 8;
        const unsigned short* gb = bRow + k0 + cb * 8;
        async_cp16(ldsA + boff,       ga);
        async_cp16(ldsA + boff + 16,  ga + 8);
        async_cp16(ldsB + boff,       gb);
        async_cp16(ldsB + boff + 16,  gb + 8);
    };

    // accumulators: [glu-half][m-frag][n-frag] as scalars (stable register homes)
    v8f ca00 = {}, ca01 = {}, ca10 = {}, ca11 = {};
    v8f cb00 = {}, cb01 = {}, cb10 = {}, cb11 = {};

    constexpr int NKT = D_ / KT;   // 64
    unsigned cur = 0;
    issue(cur, 0);
#pragma unroll 1
    for (int kt = 0; kt < NKT; ++kt) {
        unsigned nxt = cur ^ (unsigned)TILE_ELE;
        if (kt + 1 < NKT) { issue(nxt, (kt + 1) * KT); wait_async_le4(); }
        else              { wait_async_0(); }
        __syncthreads();

        const unsigned short* As = shA + cur;
        const unsigned short* Bs = shB + cur;

        v16bf a0  = lds_frag_a(As, wm * 32,      lane);
        v16bf a1  = lds_frag_a(As, wm * 32 + 16, lane);
        v16bf bA0 = lds_frag_b(Bs, wn * 32,           lane);
        v16bf bA1 = lds_frag_b(Bs, wn * 32 + 16,      lane);
        v16bf bB0 = lds_frag_b(Bs, 64 + wn * 32,      lane);
        v16bf bB1 = lds_frag_b(Bs, 64 + wn * 32 + 16, lane);

        // 8 independent WMMAs (distinct accumulators -> no D->A/B hazards)
        ca00 = wmma_bf16(a0, bA0, ca00);
        ca10 = wmma_bf16(a1, bA0, ca10);
        ca01 = wmma_bf16(a0, bA1, ca01);
        ca11 = wmma_bf16(a1, bA1, ca11);
        cb00 = wmma_bf16(a0, bB0, cb00);
        cb10 = wmma_bf16(a1, bB0, cb10);
        cb01 = wmma_bf16(a0, bB1, cb01);
        cb11 = wmma_bf16(a1, bB1, cb11);

        __syncthreads();
        cur = nxt;
    }

    // GLU epilogue: act = silu(a) * b, stored bf16
    int nl = lane & 15;
    int mh = (lane < 16) ? 0 : 8;
    int gm0 = m0 + wm * 32 + mh;
    int gn0 = n0 + wn * 32 + nl;
    glu_store(act, ca00, cb00, abase, gm0,      gn0,      cnt);
    glu_store(act, ca01, cb01, abase, gm0,      gn0 + 16, cnt);
    glu_store(act, ca10, cb10, abase, gm0 + 16, gn0,      cnt);
    glu_store(act, ca11, cb11, abase, gm0 + 16, gn0 + 16, cnt);
}

// ---------------- kernel: GEMM2 (act @ w_out^T) -> per-slot fp32 outputs ----------------
// block tile: M=128 x N=128; waves 4(M) x 2(N), wave tile 32x64
__global__ __launch_bounds__(256)
void gemm2_kernel(const unsigned short* __restrict__ act,
                  const unsigned short* __restrict__ woutb,
                  const int* __restrict__ counts, const int* __restrict__ bases,
                  float* __restrict__ obuf) {
    int e   = blockIdx.y;
    int cnt = counts[e];
    int m0  = blockIdx.z * 128;
    if (m0 >= cnt) return;
    int n0    = blockIdx.x * 128;
    int abase = bases[e];

    __shared__ unsigned short shA[2 * TILE_ELE];
    __shared__ unsigned short shB[2 * TILE_ELE];

    int tid  = threadIdx.x;
    int lane = tid & 31;
    int wave = tid >> 5;
    int wm   = wave & 3;
    int wn   = wave >> 2;

    int row = tid >> 1;
    int cb  = (tid & 1) * 2;
    int ar  = imin(m0 + row, cnt - 1);
    const unsigned short* aRow = act + (size_t)(abase + ar) * H_;
    const unsigned short* bRow = woutb + ((size_t)e * D_ + (n0 + row)) * H_;

    unsigned ldsA = (unsigned)(uintptr_t)&shA[0] + (unsigned)(row * LDST + cb * 8) * 2u;
    unsigned ldsB = (unsigned)(uintptr_t)&shB[0] + (unsigned)(row * LDST + cb * 8) * 2u;

    auto issue = [&](unsigned tileOff, int k0) {
        unsigned boff = tileOff * 2u;
        const unsigned short* ga = aRow + k0 + cb * 8;
        const unsigned short* gb = bRow + k0 + cb * 8;
        async_cp16(ldsA + boff,       ga);
        async_cp16(ldsA + boff + 16,  ga + 8);
        async_cp16(ldsB + boff,       gb);
        async_cp16(ldsB + boff + 16,  gb + 8);
    };

    v8f d00 = {}, d01 = {}, d02 = {}, d03 = {};
    v8f d10 = {}, d11 = {}, d12 = {}, d13 = {};

    constexpr int NKT = H_ / KT;   // 128
    unsigned cur = 0;
    issue(cur, 0);
#pragma unroll 1
    for (int kt = 0; kt < NKT; ++kt) {
        unsigned nxt = cur ^ (unsigned)TILE_ELE;
        if (kt + 1 < NKT) { issue(nxt, (kt + 1) * KT); wait_async_le4(); }
        else              { wait_async_0(); }
        __syncthreads();

        const unsigned short* As = shA + cur;
        const unsigned short* Bs = shB + cur;

        v16bf a0 = lds_frag_a(As, wm * 32,      lane);
        v16bf a1 = lds_frag_a(As, wm * 32 + 16, lane);
        v16bf b0 = lds_frag_b(Bs, wn * 64,      lane);
        v16bf b1 = lds_frag_b(Bs, wn * 64 + 16, lane);
        v16bf b2 = lds_frag_b(Bs, wn * 64 + 32, lane);
        v16bf b3 = lds_frag_b(Bs, wn * 64 + 48, lane);

        d00 = wmma_bf16(a0, b0, d00);
        d10 = wmma_bf16(a1, b0, d10);
        d01 = wmma_bf16(a0, b1, d01);
        d11 = wmma_bf16(a1, b1, d11);
        d02 = wmma_bf16(a0, b2, d02);
        d12 = wmma_bf16(a1, b2, d12);
        d03 = wmma_bf16(a0, b3, d03);
        d13 = wmma_bf16(a1, b3, d13);

        __syncthreads();
        cur = nxt;
    }

    int nl = lane & 15;
    int mh = (lane < 16) ? 0 : 8;
    int gm0 = m0 + wm * 32 + mh;
    int gn0 = n0 + wn * 64 + nl;
    obuf_store(obuf, d00, abase, gm0,      gn0,      cnt);
    obuf_store(obuf, d01, abase, gm0,      gn0 + 16, cnt);
    obuf_store(obuf, d02, abase, gm0,      gn0 + 32, cnt);
    obuf_store(obuf, d03, abase, gm0,      gn0 + 48, cnt);
    obuf_store(obuf, d10, abase, gm0 + 16, gn0,      cnt);
    obuf_store(obuf, d11, abase, gm0 + 16, gn0 + 16, cnt);
    obuf_store(obuf, d12, abase, gm0 + 16, gn0 + 32, cnt);
    obuf_store(obuf, d13, abase, gm0 + 16, gn0 + 48, cnt);
}

// ---------------- kernel: deterministic combine  out = bias + g0*o0 + g1*o1 ----------------
__global__ __launch_bounds__(256)
void combine_kernel(const float* __restrict__ bias, const float* __restrict__ obuf,
                    const int* __restrict__ bases, const int* __restrict__ tok_e,
                    const int* __restrict__ tok_p, const float* __restrict__ tok_g,
                    float* __restrict__ out) {
    constexpr int D4 = D_ / 4;
    for (long long i = (long long)blockIdx.x * blockDim.x + threadIdx.x;
         i < (long long)T_ * D4; i += (long long)gridDim.x * blockDim.x) {
        int t  = (int)(i / D4);
        int d4 = (int)(i % D4);
        int e0 = tok_e[2 * t],     e1 = tok_e[2 * t + 1];
        long long s0 = (long long)(bases[e0] + tok_p[2 * t])     * D4 + d4;
        long long s1 = (long long)(bases[e1] + tok_p[2 * t + 1]) * D4 + d4;
        float g0 = tok_g[2 * t], g1 = tok_g[2 * t + 1];
        float4 o0 = ((const float4*)obuf)[s0];
        float4 o1 = ((const float4*)obuf)[s1];
        float4 bv = ((const float4*)bias)[d4];
        float4 r;
        r.x = bv.x + g0 * o0.x + g1 * o1.x;
        r.y = bv.y + g0 * o0.y + g1 * o1.y;
        r.z = bv.z + g0 * o0.z + g1 * o1.z;
        r.w = bv.w + g0 * o0.w + g1 * o1.w;
        ((float4*)out)[i] = r;
    }
}

// ---------------- host launcher ----------------
extern "C" void kernel_launch(void* const* d_in, const int* in_sizes, int n_in,
                              void* d_out, int out_size, void* d_ws, size_t ws_size,
                              hipStream_t stream) {
    const float* x    = (const float*)d_in[0];   // [B,L,D]
    const float* rw   = (const float*)d_in[1];   // [E,D]
    const float* win  = (const float*)d_in[2];   // [E,2H,D]
    const float* wout = (const float*)d_in[3];   // [E,D,H]
    const float* bias = (const float*)d_in[4];   // [D]
    float* out = (float*)d_out;

    char* ws = (char*)d_ws;
    int*   counts  = (int*)(ws + OFF_COUNTS);
    int*   bases   = (int*)(ws + OFF_BASE);
    int*   tok_idx = (int*)(ws + OFF_TOKIDX);
    int*   tok_e   = (int*)(ws + OFF_TOKE);
    int*   tok_p   = (int*)(ws + OFF_TOKP);
    float* tok_g   = (float*)(ws + OFF_TOKG);
    unsigned short* xb    = (unsigned short*)(ws + OFF_XB);
    unsigned short* winb  = (unsigned short*)(ws + OFF_WINB);
    unsigned short* woutb = (unsigned short*)(ws + OFF_WOUTB);
    unsigned short* actb  = (unsigned short*)(ws + OFF_ACT);
    float*          obuf  = (float*)(ws + OFF_OBUF);

    hipMemsetAsync(counts, 0, 64, stream);

    // fp32 -> bf16 conversions
    cvt_bf16_kernel<<<4096, 256, 0, stream>>>(x,    xb,    (long long)T_ * D_ / 4);
    cvt_bf16_kernel<<<8192, 256, 0, stream>>>(win,  winb,  (long long)E_ * H2_ * D_ / 4);
    cvt_bf16_kernel<<<8192, 256, 0, stream>>>(wout, woutb, (long long)E_ * D_ * H_ / 4);

    // routing + dispatch
    router_kernel<<<T_ / 8, 256, 0, stream>>>(x, rw, counts, tok_idx, tok_e, tok_p, tok_g);
    scan_kernel<<<1, 32, 0, stream>>>(counts, bases);

    // expert GEMM1 + GLU  (grid: n-tiles x experts x max m-tiles)
    gemm1_glu_kernel<<<dim3(H_ / 64, E_, T_ / 128), 256, 0, stream>>>(
        xb, winb, counts, bases, tok_idx, actb);

    // expert GEMM2
    gemm2_kernel<<<dim3(D_ / 128, E_, T_ / 128), 256, 0, stream>>>(
        actb, woutb, counts, bases, obuf);

    // deterministic combine with bias
    combine_kernel<<<4096, 256, 0, stream>>>(bias, obuf, bases, tok_e, tok_p, tok_g, out);

    (void)in_sizes; (void)n_in; (void)out_size; (void)ws_size;
}
